// SelfAttn_10909216932104
// MI455X (gfx1250) — matrix operand order repaired and checked by reading, not compiled, not run on hardware
//
#include <hip/hip_runtime.h>

// ---------------------------------------------------------------------------
// MI455X (gfx1250) fused SAGAN self-attention.
// wave32, WMMA bf16 (v_wmma_f32_16x16x32_bf16), flash-style fused softmax.
// Round 3: cooperative tile staging now uses gfx1250 async direct-to-LDS
// loads (global_load_async_to_lds_b128 + s_wait_asynccnt) instead of the
// global->VGPR->ds_store round trip.
// ---------------------------------------------------------------------------

typedef __bf16 bf16_t;
typedef __attribute__((ext_vector_type(16))) __bf16 v16bf;
typedef __attribute__((ext_vector_type(4)))  __bf16 v4bf;
typedef __attribute__((ext_vector_type(8)))  float  v8f;

#define WMMA_BF16(a, b, c) \
  __builtin_amdgcn_wmma_f32_16x16x32_bf16(false, (a), false, (b), (short)0, (c), false, false)

// Problem constants
#define NB   8
#define CC   512
#define CK   64
#define CV   256
#define HW   4096            // 64*64
#define MM   32768           // NB*HW
#define PP   1024            // HW/4

// --- async direct-to-LDS 16B load (ASYNCcnt-tracked) ------------------------
// Generic pointers to __shared__ carry the LDS byte offset in their low 32
// bits (ISA 10.2 aperture mapping), which is what VDST wants.
__device__ __forceinline__ uint32_t lds_off32(const void* p) {
  return (uint32_t)(uintptr_t)p;
}
__device__ __forceinline__ void async_g2l_b128(uint32_t lds_off, const void* gaddr) {
  asm volatile("global_load_async_to_lds_b128 %0, %1, off"
               :: "v"(lds_off), "v"(gaddr)
               : "memory");
}
__device__ __forceinline__ void wait_async0() {
  asm volatile("s_wait_asynccnt 0" ::: "memory");
}

// --- fragment loaders -------------------------------------------------------
// A (16x32 bf16, MxK): lane row = lane%16; h = lane/16.
//   elems[0..7]  = A[row][k0 + h*8 + j]
//   elems[8..15] = A[row][k0 + 16 + h*8 + j]
__device__ __forceinline__ v16bf load_a_frag(const bf16_t* row, int k0, int h) {
  union { v16bf v; uint4 u[2]; } r;
  r.u[0] = *(const uint4*)(row + k0 + h * 8);
  r.u[1] = *(const uint4*)(row + k0 + 16 + h * 8);
  return r.v;
}
// B (32x16 bf16, KxN) stored transposed as Bt[N][K]: lane col = lane%16.
//   elems[0..7]  = Bt[col][k0 + h*8 + j]
//   elems[8..15] = Bt[col][k0 + 16 + h*8 + j]
__device__ __forceinline__ v16bf load_b_frag(const bf16_t* rowN, int k0, int h) {
  union { v16bf v; uint4 u[2]; } r;
  r.u[0] = *(const uint4*)(rowN + k0 + h * 8);
  r.u[1] = *(const uint4*)(rowN + k0 + 16 + h * 8);
  return r.v;
}

__device__ __forceinline__ float half_max(float x) {
  x = fmaxf(x, __shfl_xor(x, 1, 16));
  x = fmaxf(x, __shfl_xor(x, 2, 16));
  x = fmaxf(x, __shfl_xor(x, 4, 16));
  x = fmaxf(x, __shfl_xor(x, 8, 16));
  return x;
}
__device__ __forceinline__ float half_sum(float x) {
  x += __shfl_xor(x, 1, 16);
  x += __shfl_xor(x, 2, 16);
  x += __shfl_xor(x, 4, 16);
  x += __shfl_xor(x, 8, 16);
  return x;
}

// --- 1) x -> bf16 -----------------------------------------------------------
__global__ void cvt_x_kernel(const float* __restrict__ x, bf16_t* __restrict__ xbf, int n4) {
  int i = blockIdx.x * blockDim.x + threadIdx.x;
  if (i < n4) {
    float4 f = ((const float4*)x)[i];
    v4bf o;
    o[0] = (bf16_t)f.x; o[1] = (bf16_t)f.y; o[2] = (bf16_t)f.z; o[3] = (bf16_t)f.w;
    ((v4bf*)xbf)[i] = o;
  }
}

// --- 2) W[K][N] -> Wt[N][K] bf16 -------------------------------------------
__global__ void transpose_cvt_kernel(const float* __restrict__ in, bf16_t* __restrict__ outT,
                                     int K, int N) {
  int i = blockIdx.x * blockDim.x + threadIdx.x;  // i = n*K + k
  if (i < N * K) {
    int n = i / K, k = i % K;
    outT[i] = (bf16_t)in[(size_t)k * N + n];
  }
}

// --- 3a) theta + phi projections (8 col tiles, 64 acc VGPRs/wave) -----------
// grid 512 x 128 thr; wave = 16 rows x 128 cols; K = 512.
__global__ __launch_bounds__(128) void proj_tp_kernel(
    const bf16_t* __restrict__ xbf,
    const bf16_t* __restrict__ wThT, const bf16_t* __restrict__ wPhT,
    bf16_t* __restrict__ thetaBf, float* __restrict__ phiF) {
  __shared__ bf16_t Wlds[128 * 32];  // 8 KB slab: 128 cols x 32 k
  const int tid = threadIdx.x;
  const int wid = tid >> 5, lane = tid & 31;
  const int h = lane >> 4, n16 = lane & 15;
  const int r0 = (blockIdx.x * 4 + wid) * 16;
  const bf16_t* xrow = xbf + (size_t)(r0 + n16) * CC;

  v8f acc[8];
  const v8f zero = {0.f, 0.f, 0.f, 0.f, 0.f, 0.f, 0.f, 0.f};
#pragma unroll
  for (int t = 0; t < 8; ++t) acc[t] = zero;

  for (int s = 0; s < 16; ++s) {  // k = 32*s
    __syncthreads();
#pragma unroll
    for (int c0 = 0; c0 < 4; ++c0) {  // 512 16B chunks, 4 per thread
      int c = tid + c0 * 128;
      int n = c >> 2, part = c & 3;
      const bf16_t* src = (n < 64) ? (wThT + (size_t)n * CC) : (wPhT + (size_t)(n - 64) * CC);
      async_g2l_b128(lds_off32(&Wlds[n * 32 + part * 8]), src + s * 32 + part * 8);
    }
    wait_async0();
    __syncthreads();
    v16bf a = load_a_frag(xrow, s * 32, h);
    v16bf b = load_b_frag(&Wlds[n16 * 32], 0, h);
#pragma unroll
    for (int t = 0; t < 8; ++t) {
      v16bf bn = b;
      if (t + 1 < 8) bn = load_b_frag(&Wlds[((t + 1) * 16 + n16) * 32], 0, h);
      acc[t] = WMMA_BF16(a, b, acc[t]);
      b = bn;
    }
  }
#pragma unroll
  for (int t = 0; t < 8; ++t) {
#pragma unroll
    for (int v = 0; v < 8; ++v) {
      int m = v + 8 * h;
      size_t row = (size_t)(r0 + m);
      float val = acc[t][v];
      if (t < 4) thetaBf[row * CK + t * 16 + n16] = (bf16_t)val;
      else       phiF[row * CK + (t - 4) * 16 + n16] = val;
    }
  }
}

// --- 3b) g projection (16 col tiles, 128 acc VGPRs/wave) --------------------
__global__ __launch_bounds__(128) void proj_g_kernel(
    const bf16_t* __restrict__ xbf, const bf16_t* __restrict__ wGT,
    float* __restrict__ gF) {
  __shared__ bf16_t Wlds[256 * 32];  // 16 KB slab
  const int tid = threadIdx.x;
  const int wid = tid >> 5, lane = tid & 31;
  const int h = lane >> 4, n16 = lane & 15;
  const int r0 = (blockIdx.x * 4 + wid) * 16;
  const bf16_t* xrow = xbf + (size_t)(r0 + n16) * CC;

  v8f acc[16];
  const v8f zero = {0.f, 0.f, 0.f, 0.f, 0.f, 0.f, 0.f, 0.f};
#pragma unroll
  for (int t = 0; t < 16; ++t) acc[t] = zero;

  for (int s = 0; s < 16; ++s) {
    __syncthreads();
#pragma unroll
    for (int c0 = 0; c0 < 8; ++c0) {  // 1024 chunks
      int c = tid + c0 * 128;
      int n = c >> 2, part = c & 3;
      async_g2l_b128(lds_off32(&Wlds[n * 32 + part * 8]),
                     wGT + (size_t)n * CC + s * 32 + part * 8);
    }
    wait_async0();
    __syncthreads();
    v16bf a = load_a_frag(xrow, s * 32, h);
    v16bf b = load_b_frag(&Wlds[n16 * 32], 0, h);
#pragma unroll
    for (int t = 0; t < 16; ++t) {
      v16bf bn = b;
      if (t + 1 < 16) bn = load_b_frag(&Wlds[((t + 1) * 16 + n16) * 32], 0, h);
      acc[t] = WMMA_BF16(a, b, acc[t]);
      b = bn;
    }
  }
#pragma unroll
  for (int t = 0; t < 16; ++t) {
#pragma unroll
    for (int v = 0; v < 8; ++v) {
      int m = v + 8 * h;
      gF[(size_t)(r0 + m) * CV + t * 16 + n16] = acc[t][v];
    }
  }
}

// --- 4) 2x2 maxpool ---------------------------------------------------------
__global__ void pool_phi_kernel(const float* __restrict__ phiF, bf16_t* __restrict__ phiP) {
  int i = blockIdx.x * blockDim.x + threadIdx.x;  // [NB][PP][CK]
  int n = i & 63, p = (i >> 6) & 1023, b = i >> 16;
  int ph = p >> 5, pw = p & 31;
  size_t base = (size_t)b * HW + ph * 128 + pw * 2;
  float v0 = phiF[(base) * CK + n];
  float v1 = phiF[(base + 1) * CK + n];
  float v2 = phiF[(base + 64) * CK + n];
  float v3 = phiF[(base + 65) * CK + n];
  phiP[i] = (bf16_t)fmaxf(fmaxf(v0, v1), fmaxf(v2, v3));
}
// g pooled AND transposed: gPT[b][v][p] (V^T is the B-operand of P@V)
__global__ void pool_g_kernel(const float* __restrict__ gF, bf16_t* __restrict__ gPT) {
  int i = blockIdx.x * blockDim.x + threadIdx.x;  // [NB][CV][PP]
  int p = i & 1023, v = (i >> 10) & 255, b = i >> 18;
  int ph = p >> 5, pw = p & 31;
  size_t base = (size_t)b * HW + ph * 128 + pw * 2;
  float v0 = gF[(base) * CV + v];
  float v1 = gF[(base + 1) * CV + v];
  float v2 = gF[(base + 64) * CV + v];
  float v3 = gF[(base + 65) * CV + v];
  gPT[i] = (bf16_t)fmaxf(fmaxf(v0, v1), fmaxf(v2, v3));
}

// --- 5) fused flash attention: ag = softmax(theta phi^T) g ------------------
// grid (NB*64) x 128 thr; wave = 16 queries; streams 1024 keys in 32-tiles.
__global__ __launch_bounds__(128) void attn_kernel(
    const bf16_t* __restrict__ thetaBf, const bf16_t* __restrict__ phiP,
    const bf16_t* __restrict__ gPT, bf16_t* __restrict__ agBf) {
  __shared__ char smem[32768];
  bf16_t* Klds = (bf16_t*)smem;                       // [32][64]  (4 KB)
  bf16_t* Vlds = (bf16_t*)(smem + 4096);              // [256][32] (16 KB)
  bf16_t* PldsAll = (bf16_t*)(smem + 4096 + 16384);   // [4][16][32] (4 KB)

  const int tid = threadIdx.x;
  const int wid = tid >> 5, lane = tid & 31;
  const int h = lane >> 4, n16 = lane & 15;
  const int batch = blockIdx.x >> 6;
  const int q0 = (blockIdx.x & 63) * 64 + wid * 16;

  const bf16_t* qrow = thetaBf + ((size_t)batch * HW + q0 + n16) * CK;
  v16bf aQ0 = load_a_frag(qrow, 0, h);
  v16bf aQ1 = load_a_frag(qrow, 32, h);
  bf16_t* Pw = PldsAll + wid * 512;

  const v8f zero = {0.f, 0.f, 0.f, 0.f, 0.f, 0.f, 0.f, 0.f};
  v8f O[16];
  float mrow[8], lrow[8];
#pragma unroll
  for (int t = 0; t < 16; ++t) O[t] = zero;
#pragma unroll
  for (int v = 0; v < 8; ++v) { mrow[v] = -1e30f; lrow[v] = 0.f; }

  for (int it = 0; it < 32; ++it) {
    const int p0 = it * 32;
    __syncthreads();  // previous iteration fully consumed K/V before overwrite
#pragma unroll
    for (int c0 = 0; c0 < 10; ++c0) {  // 1280 16B chunks, 10 per thread
      int c = tid + c0 * 128;
      if (c < 256) {  // K tile [32 p][64 k]
        int row = c >> 3, part = c & 7;
        async_g2l_b128(lds_off32(&Klds[row * 64 + part * 8]),
                       phiP + ((size_t)batch * PP + p0 + row) * CK + part * 8);
      } else {        // V^T tile [256 v][32 p]
        int cv = c - 256, v = cv >> 2, part = cv & 3;
        async_g2l_b128(lds_off32(&Vlds[v * 32 + part * 8]),
                       gPT + ((size_t)batch * CV + v) * PP + p0 + part * 8);
      }
    }
    wait_async0();
    __syncthreads();

    // S = Q K^T : [16 q x 32 p]
    v8f S0 = zero, S1 = zero;
    {
      v16bf b0 = load_b_frag(&Klds[(n16) * 64], 0, h);
      v16bf b1 = load_b_frag(&Klds[(n16) * 64], 32, h);
      v16bf b2 = load_b_frag(&Klds[(16 + n16) * 64], 0, h);
      v16bf b3 = load_b_frag(&Klds[(16 + n16) * 64], 32, h);
      S0 = WMMA_BF16(aQ0, b0, S0);
      S0 = WMMA_BF16(aQ1, b1, S0);
      S1 = WMMA_BF16(aQ0, b2, S1);
      S1 = WMMA_BF16(aQ1, b3, S1);
    }

    // online softmax over the 32 new columns
#pragma unroll
    for (int v = 0; v < 8; ++v) {
      float rx = half_max(fmaxf(S0[v], S1[v]));
      float mnew = fmaxf(mrow[v], rx);
      float corr = __expf(mrow[v] - mnew);
      float e0 = __expf(S0[v] - mnew);
      float e1 = __expf(S1[v] - mnew);
      lrow[v] = lrow[v] * corr + half_sum(e0 + e1);
      mrow[v] = mnew;
#pragma unroll
      for (int t = 0; t < 16; ++t) O[t][v] *= corr;
      int m = v + 8 * h;
      Pw[m * 32 + n16] = (bf16_t)e0;
      Pw[m * 32 + 16 + n16] = (bf16_t)e1;
    }
    asm volatile("s_wait_dscnt 0" ::: "memory");  // wave-local LDS RAW

    // O += P V : K = 32 (this p-tile), pipelined B loads
    v16bf aP = load_a_frag(Pw + n16 * 32, 0, h);
    v16bf bV = load_b_frag(&Vlds[n16 * 32], 0, h);
#pragma unroll
    for (int t = 0; t < 16; ++t) {
      v16bf bn = bV;
      if (t + 1 < 16) bn = load_b_frag(&Vlds[((t + 1) * 16 + n16) * 32], 0, h);
      O[t] = WMMA_BF16(aP, bV, O[t]);
      bV = bn;
    }
  }

  // normalize
#pragma unroll
  for (int v = 0; v < 8; ++v) {
    float inv = 1.0f / lrow[v];
#pragma unroll
    for (int t = 0; t < 16; ++t) O[t][v] *= inv;
  }

  // stage per-wave [16][256] bf16 in LDS, then coalesced b128 stores
  __syncthreads();
  bf16_t* ost = (bf16_t*)smem + (size_t)wid * 4096;
#pragma unroll
  for (int t = 0; t < 16; ++t) {
#pragma unroll
    for (int v = 0; v < 8; ++v) {
      int m = v + 8 * h;
      ost[m * 256 + t * 16 + n16] = (bf16_t)O[t][v];
    }
  }
  asm volatile("s_wait_dscnt 0" ::: "memory");
  bf16_t* agBase = agBf + ((size_t)batch * HW + q0) * CV;
  for (int r = 0; r < 16; ++r) {
    uint4 d = *(const uint4*)(ost + r * 256 + lane * 8);
    *(uint4*)(agBase + (size_t)r * 256 + lane * 8) = d;
  }
}

// --- 6) out = x + gamma * (ag @ w_o) ----------------------------------------
// grid (512, 2) x 128 thr; wave = 16 rows x 256 cols (col half via blockIdx.y)
__global__ __launch_bounds__(128) void out_kernel(
    const bf16_t* __restrict__ agBf, const bf16_t* __restrict__ wOT,
    const float* __restrict__ x, const float* __restrict__ gammaPtr,
    float* __restrict__ out) {
  __shared__ bf16_t Wlds[256 * 32];  // 16 KB
  const int tid = threadIdx.x;
  const int wid = tid >> 5, lane = tid & 31;
  const int h = lane >> 4, n16 = lane & 15;
  const int r0 = (blockIdx.x * 4 + wid) * 16;
  const int colbase = blockIdx.y * 256;
  const float gamma = gammaPtr[0];
  const bf16_t* arow = agBf + (size_t)(r0 + n16) * CV;

  v8f acc[16];
  const v8f zero = {0.f, 0.f, 0.f, 0.f, 0.f, 0.f, 0.f, 0.f};
#pragma unroll
  for (int t = 0; t < 16; ++t) acc[t] = zero;

  for (int s = 0; s < 8; ++s) {  // K = 256
    __syncthreads();
#pragma unroll
    for (int c0 = 0; c0 < 8; ++c0) {  // 1024 16B chunks
      int c = tid + c0 * 128;
      int n = c >> 2, part = c & 3;
      async_g2l_b128(lds_off32(&Wlds[n * 32 + part * 8]),
                     wOT + (size_t)(colbase + n) * CV + s * 32 + part * 8);
    }
    wait_async0();
    __syncthreads();
    v16bf a = load_a_frag(arow, s * 32, h);
    v16bf b = load_b_frag(&Wlds[n16 * 32], 0, h);
#pragma unroll
    for (int t = 0; t < 16; ++t) {
      v16bf bn = b;
      if (t + 1 < 16) bn = load_b_frag(&Wlds[((t + 1) * 16 + n16) * 32], 0, h);
      acc[t] = WMMA_BF16(a, b, acc[t]);
      b = bn;
    }
  }
#pragma unroll
  for (int t = 0; t < 16; ++t) {
#pragma unroll
    for (int v = 0; v < 8; ++v) {
      int m = v + 8 * h;
      size_t idx = (size_t)(r0 + m) * CC + colbase + t * 16 + n16;
      out[idx] = x[idx] + gamma * acc[t][v];
    }
  }
}

// ---------------------------------------------------------------------------
extern "C" void kernel_launch(void* const* d_in, const int* in_sizes, int n_in,
                              void* d_out, int out_size, void* d_ws, size_t ws_size,
                              hipStream_t stream) {
  const float* x = (const float*)d_in[0];
  const float* w_theta = (const float*)d_in[1];
  const float* w_phi = (const float*)d_in[2];
  const float* w_g = (const float*)d_in[3];
  const float* w_o = (const float*)d_in[4];
  const float* gamma = (const float*)d_in[5];
  float* out = (float*)d_out;

  char* Wb = (char*)d_ws;
  size_t off = 0;
  auto take = [&](size_t bytes) { size_t o = off; off += (bytes + 255) & ~(size_t)255; return o; };
  bf16_t* x_bf    = (bf16_t*)(Wb + take((size_t)MM * CC * 2));   // 32 MB
  bf16_t* thetaBf = (bf16_t*)(Wb + take((size_t)MM * CK * 2));   // 4 MB
  float*  phiF    = (float*)(Wb + take((size_t)MM * CK * 4));    // 8 MB
  float*  gF      = (float*)(Wb + take((size_t)MM * CV * 4));    // 32 MB
  bf16_t* phiP    = (bf16_t*)(Wb + take((size_t)NB * PP * CK * 2));  // 1 MB
  bf16_t* gPT     = (bf16_t*)(Wb + take((size_t)NB * CV * PP * 2));  // 4 MB
  bf16_t* agBf    = (bf16_t*)(Wb + take((size_t)MM * CV * 2));   // 16 MB
  bf16_t* wThT    = (bf16_t*)(Wb + take((size_t)CK * CC * 2));
  bf16_t* wPhT    = (bf16_t*)(Wb + take((size_t)CK * CC * 2));
  bf16_t* wGT     = (bf16_t*)(Wb + take((size_t)CV * CC * 2));
  bf16_t* wOT     = (bf16_t*)(Wb + take((size_t)CC * CV * 2));

  cvt_x_kernel<<<(MM * CC / 4) / 256, 256, 0, stream>>>(x, x_bf, MM * CC / 4);
  transpose_cvt_kernel<<<(CC * CK) / 256, 256, 0, stream>>>(w_theta, wThT, CC, CK);
  transpose_cvt_kernel<<<(CC * CK) / 256, 256, 0, stream>>>(w_phi, wPhT, CC, CK);
  transpose_cvt_kernel<<<(CC * CV) / 256, 256, 0, stream>>>(w_g, wGT, CC, CV);
  transpose_cvt_kernel<<<(CV * CC) / 256, 256, 0, stream>>>(w_o, wOT, CV, CC);

  proj_tp_kernel<<<MM / 64, 128, 0, stream>>>(x_bf, wThT, wPhT, thetaBf, phiF);
  proj_g_kernel<<<MM / 64, 128, 0, stream>>>(x_bf, wGT, gF);
  pool_phi_kernel<<<(NB * PP * CK) / 256, 256, 0, stream>>>(phiF, phiP);
  pool_g_kernel<<<(NB * CV * PP) / 256, 256, 0, stream>>>(gF, gPT);

  attn_kernel<<<NB * 64, 128, 0, stream>>>(thetaBf, phiP, gPT, agBf);

  out_kernel<<<dim3(MM / 64, 2), 128, 0, stream>>>(agBf, wOT, x, gamma, out);
}
